// T5Attention_25477745999877
// MI455X (gfx1250) — matrix operand order repaired and checked by reading, not compiled, not run on hardware
//
#include <hip/hip_runtime.h>
#include <hip/hip_bf16.h>

#define NB 2
#define LL 2048
#define DMODEL 768
#define NHEADS 12
#define HDIM 64

typedef __bf16 bf16_t;
typedef bf16_t v16bf __attribute__((ext_vector_type(16)));
typedef float v8f __attribute__((ext_vector_type(8)));
typedef unsigned short u16;
typedef unsigned int u32;
typedef u16 u16x8 __attribute__((ext_vector_type(8)));
typedef u16 u16x16 __attribute__((ext_vector_type(16)));
typedef u32 u32x4 __attribute__((ext_vector_type(4)));
typedef int i32x4 __attribute__((ext_vector_type(4)));
typedef int i32x8 __attribute__((ext_vector_type(8)));

#if __has_builtin(__builtin_amdgcn_tensor_load_to_lds) && \
    __has_builtin(__builtin_amdgcn_s_wait_tensorcnt)
#define HAVE_TDM 1
#else
#define HAVE_TDM 0
#endif

union V16U { v16bf bf; u16x16 us; };

// f32 -> bf16 (round to nearest even), pure bit ops
__device__ __forceinline__ u16 f2bf(float f) {
  unsigned u = __builtin_bit_cast(unsigned, f);
  u += 0x7fffu + ((u >> 16) & 1u);
  return (u16)(u >> 16);
}

// Build a 16-element bf16 fragment from two 8-element (16-byte) loads.
__device__ __forceinline__ v16bf ld8x2(const u16* p0, const u16* p1) {
  V16U r;
  u16x8 lo = *(const u16x8*)p0;
  u16x8 hi = *(const u16x8*)p1;
#pragma unroll
  for (int i = 0; i < 8; ++i) { r.us[i] = lo[i]; r.us[i + 8] = hi[i]; }
  return r.bf;
}

__device__ __forceinline__ v8f wmma_bf16(v16bf a, v16bf b, v8f c) {
  return __builtin_amdgcn_wmma_f32_16x16x32_bf16(false, a, false, b,
                                                 (short)0, c, false, false);
}

#if HAVE_TDM
// Issue a TDM 2-D tile load (bf16 elements) into LDS.
// D# per CDNA5 ISA 8.3/8.4: group0 = {flags, lds_addr, global_addr, type},
// group1 = {data_size, tensor dims/strides, tile dims}.  Dims/strides in
// element (data_size) units; addresses in bytes.
// This toolchain exposes the 6-arg builtin:
//   (u32x4 g0, i32x8 g1, i32x4 g2, i32x4 g3, i32x8 extra, i32 cpol)
__device__ __forceinline__ void tdm_load_2d(u32 lds_off, const void* gaddr,
                                            u32 tdim0, u32 tdim1, u32 stride0,
                                            u32 tile0, u32 tile1) {
  unsigned long long ga = (unsigned long long)(uintptr_t)gaddr;
  u32x4 g0;
  g0[0] = 1u;                                            // count=1, user mode
  g0[1] = lds_off;                                       // LDS byte address
  g0[2] = (u32)ga;                                       // global_addr[31:0]
  g0[3] = (u32)((ga >> 32) & 0x01FFFFFFu) | (2u << 30);  // addr[56:32]|type=2
  i32x8 g1;
  g1[0] = (int)(1u << 16);                               // data_size = 2 bytes
  g1[1] = (int)((tdim0 & 0xFFFFu) << 16);                // tensor_dim0[15:0]
  g1[2] = (int)((tdim0 >> 16) | ((tdim1 & 0xFFFFu) << 16));
  g1[3] = (int)((tdim1 >> 16) | (tile0 << 16));          // tile_dim0
  g1[4] = (int)(tile1 & 0xFFFFu);                        // tile_dim1, dim2=0
  g1[5] = (int)stride0;                                  // dim0 stride lo32
  g1[6] = 0;
  g1[7] = 0;
  i32x4 z4 = {};
  i32x8 z8 = {};
  __builtin_amdgcn_tensor_load_to_lds(g0, g1, z4, z4, z8, 0);
}
#endif

// ---------------------------------------------------------------------------
// Kernel 1: Q/K/V projection.  D = X @ W^T, bf16 outputs.
//   TRANSPOSED=false -> [b,h,L,64]   (Q, K)
//   TRANSPOSED=true  -> [b,h,64,L]   (V^T, so P@V B-fragments are contiguous)
// One wave per 16x16 output tile; K-loop over 768 in steps of 32.
// ---------------------------------------------------------------------------
template <bool TRANSPOSED>
__global__ void __launch_bounds__(128)
qkv_proj_kernel(const float* __restrict__ x, const float* __restrict__ W,
                u16* __restrict__ dst) {
  const int lane = threadIdx.x;
  const int r = lane & 15, half = lane >> 4;
  const int m0 = (blockIdx.x * 4 + threadIdx.y) * 16;
  const int n0 = blockIdx.y * 16;

  const float* xrow = x + (size_t)(m0 + r) * DMODEL;
  const float* wrow = W + (size_t)(n0 + r) * DMODEL;

  v8f acc = {};
  for (int k0 = 0; k0 < DMODEL; k0 += 32) {
    V16U a, bw;
    const float* xa = xrow + k0 + half * 16;  // A: row m, K slices per layout
#pragma unroll
    for (int j = 0; j < 8; ++j) { a.us[j] = f2bf(xa[j]); a.us[j + 8] = f2bf(xa[8 + j]); }
    const float* wb = wrow + k0 + half * 16;  // B column n == W row n (x@W^T)
#pragma unroll
    for (int j = 0; j < 16; ++j) bw.us[j] = f2bf(wb[j]);
    acc = wmma_bf16(a.bf, bw.bf, acc);
  }

#pragma unroll
  for (int i = 0; i < 8; ++i) {
    int m = m0 + half * 8 + i;            // C/D: VGPR i -> row 8*half+i
    int n = n0 + r;                       //      lane&15 -> column
    int bb = m >> 11, l = m & (LL - 1);
    int h = n >> 6, d = n & (HDIM - 1);
    u16 val = f2bf(acc[i]);
    if (TRANSPOSED)
      dst[((size_t)(bb * NHEADS + h) * HDIM + d) * LL + l] = val;
    else
      dst[((size_t)(bb * NHEADS + h) * LL + l) * HDIM + d] = val;
  }
}

// ---------------------------------------------------------------------------
// Kernel 2: flash attention.  4 waves/block share one (b,h); each wave owns a
// 16-row query tile.  Per 32-key chunk the K tile (32x64) and V^T tile (64x32)
// are staged once into LDS (TDM when available), then all 4 waves consume
// them.  Online softmax in f32; P->A-layout via per-wave LDS tile.
// ---------------------------------------------------------------------------
__global__ void __launch_bounds__(128)
attn_kernel(const u16* __restrict__ q, const u16* __restrict__ kmat,
            const u16* __restrict__ vt, const float* __restrict__ pos_bias,
            const int* __restrict__ mask, u16* __restrict__ ao) {
  __shared__ __align__(16) u16 kbuf[32 * HDIM];      // K rows kc..kc+31
  __shared__ __align__(16) u16 vbuf[HDIM * 32];      // V^T rows 0..63, 32 cols
  __shared__ __align__(16) u16 plds[4][16 * 32];

  const int lane = threadIdx.x;
  const int r = lane & 15, half = lane >> 4;
  const int wave = threadIdx.y;
  const int q0 = (blockIdx.x * 4 + wave) * 16;
  const int h = blockIdx.y;
  const int b = blockIdx.z;

  const u16* qb = q + (size_t)(b * NHEADS + h) * LL * HDIM;
  const u16* kb = kmat + (size_t)(b * NHEADS + h) * LL * HDIM;
  const u16* vb = vt + (size_t)(b * NHEADS + h) * HDIM * LL;
  const float* biasb = pos_bias + ((size_t)h * LL + q0) * LL;  // [1,NH,L,L]
  const int* mkb = mask + b * LL;
  u16* pl = plds[wave];

#if HAVE_TDM
  const u32 kbuf_off = (u32)(uintptr_t)(void*)kbuf;  // LDS aperture: addr[31:0]
  const u32 vbuf_off = (u32)(uintptr_t)(void*)vbuf;
#endif

  // Q A-fragments for HD 0..31 and 32..63 (loaded once)
  const u16* qr = qb + (size_t)(q0 + r) * HDIM;
  v16bf aq0 = ld8x2(qr + half * 16, qr + half * 16 + 8);
  v16bf aq1 = ld8x2(qr + 32 + half * 16, qr + 32 + half * 16 + 8);

  float mrow[8], lrow[8];
  v8f oacc[4] = {};
#pragma unroll
  for (int i = 0; i < 8; ++i) { mrow[i] = -3.0e38f; lrow[i] = 0.0f; }

  for (int kc = 0; kc < LL; kc += 32) {
    // ---- stage K / V^T chunk into LDS, once per block ---------------------
#if HAVE_TDM
    if (wave == 0) {
      // K:  tensor [2048 x 64], tile rows kc..kc+31, all 64 cols
      tdm_load_2d(kbuf_off, kb + (size_t)kc * HDIM, HDIM, LL, HDIM, HDIM, 32);
      // V^T: tensor [64 x 2048], tile cols kc..kc+31, all 64 rows
      tdm_load_2d(vbuf_off, vb + kc, LL, HDIM, LL, 32, HDIM);
      __builtin_amdgcn_s_wait_tensorcnt((short)0);
    }
#else
    {
      int tid = wave * 32 + lane;                      // 0..127
      const u16* kg = kb + (size_t)kc * HDIM;          // 2048 u16 total
      *(u16x8*)(kbuf + tid * 16)     = *(const u16x8*)(kg + tid * 16);
      *(u16x8*)(kbuf + tid * 16 + 8) = *(const u16x8*)(kg + tid * 16 + 8);
      int d = tid >> 1, t = (tid & 1) * 16;            // 64 rows x 32 cols
      const u16* vg = vb + (size_t)d * LL + kc + t;
      *(u16x8*)(vbuf + d * 32 + t)     = *(const u16x8*)vg;
      *(u16x8*)(vbuf + d * 32 + t + 8) = *(const u16x8*)(vg + 8);
    }
#endif
    __syncthreads();

    // ---- scores: two 16x16 tiles covering 32 keys -------------------------
    v8f s0 = {}, s1 = {};
    {
      const u16* krow = kbuf + (size_t)r * HDIM;       // B column j = K row
      v16bf bk0 = ld8x2(krow + half * 16, krow + half * 16 + 8);
      v16bf bk1 = ld8x2(krow + 32 + half * 16, krow + 32 + half * 16 + 8);
      s0 = wmma_bf16(aq0, bk0, s0);
      s0 = wmma_bf16(aq1, bk1, s0);
    }
    {
      const u16* krow = kbuf + (size_t)(16 + r) * HDIM;
      v16bf bk0 = ld8x2(krow + half * 16, krow + half * 16 + 8);
      v16bf bk1 = ld8x2(krow + 32 + half * 16, krow + 32 + half * 16 + 8);
      s1 = wmma_bf16(aq0, bk0, s1);
      s1 = wmma_bf16(aq1, bk1, s1);
    }

    // ---- additive bias + key mask ----------------------------------------
    float madd0 = (mkb[kc + r] == 0) ? -3.0e38f : 0.0f;
    float madd1 = (mkb[kc + 16 + r] == 0) ? -3.0e38f : 0.0f;
    const float* brow = biasb + (size_t)(half * 8) * LL + kc;
    __builtin_prefetch(brow + 32, 0, 0);   // next chunk's bias cacheline
#pragma unroll
    for (int i = 0; i < 8; ++i) {
      s0[i] += brow[(size_t)i * LL + r] + madd0;
      s1[i] += brow[(size_t)i * LL + 16 + r] + madd1;
    }

    // ---- online softmax (rows live in half-groups of 16 lanes) -----------
    float mx[8];
#pragma unroll
    for (int i = 0; i < 8; ++i) mx[i] = fmaxf(s0[i], s1[i]);
#pragma unroll
    for (int off = 8; off >= 1; off >>= 1)
#pragma unroll
      for (int i = 0; i < 8; ++i) mx[i] = fmaxf(mx[i], __shfl_xor(mx[i], off, 32));

    float alpha[8], rs[8];
#pragma unroll
    for (int i = 0; i < 8; ++i) {
      float mnew = fmaxf(mrow[i], mx[i]);
      alpha[i] = __expf(mrow[i] - mnew);
      mrow[i] = mnew;
      float p0 = __expf(s0[i] - mnew);
      float p1 = __expf(s1[i] - mnew);
      s0[i] = p0; s1[i] = p1;
      rs[i] = p0 + p1;
    }
#pragma unroll
    for (int off = 8; off >= 1; off >>= 1)
#pragma unroll
      for (int i = 0; i < 8; ++i) rs[i] += __shfl_xor(rs[i], off, 32);
#pragma unroll
    for (int i = 0; i < 8; ++i) lrow[i] = lrow[i] * alpha[i] + rs[i];
#pragma unroll
    for (int n = 0; n < 4; ++n)
#pragma unroll
      for (int i = 0; i < 8; ++i) oacc[n][i] *= alpha[i];

    // ---- P: C/D layout -> LDS -> A-fragment layout -----------------------
#pragma unroll
    for (int i = 0; i < 8; ++i) {
      int row = half * 8 + i;
      pl[row * 32 + r] = f2bf(s0[i]);
      pl[row * 32 + 16 + r] = f2bf(s1[i]);
    }
    __syncthreads();
    v16bf ap = ld8x2(pl + r * 32 + half * 16, pl + r * 32 + half * 16 + 8);

    // ---- O += P @ V  (V^T rows staged in LDS, contiguous B columns) ------
#pragma unroll
    for (int n = 0; n < 4; ++n) {
      const u16* vcol = vbuf + (size_t)(n * 16 + r) * 32 + half * 16;
      v16bf bv = ld8x2(vcol, vcol + 8);
      oacc[n] = wmma_bf16(ap, bv, oacc[n]);
    }
    __syncthreads();   // all waves done with kbuf/vbuf before next stage
  }

  // ---- normalize and emit [B, L, NH*HD] bf16 -----------------------------
#pragma unroll
  for (int i = 0; i < 8; ++i) {
    float inv = 1.0f / lrow[i];
    int m = q0 + half * 8 + i;
    u16* dst = ao + (size_t)(b * LL + m) * DMODEL + h * HDIM;
#pragma unroll
    for (int n = 0; n < 4; ++n) dst[n * 16 + r] = f2bf(oacc[n][i] * inv);
  }
}

// ---------------------------------------------------------------------------
// Kernel 3: output projection  out = AO @ Wo^T  (bf16 x bf16 -> f32)
// ---------------------------------------------------------------------------
__global__ void __launch_bounds__(128)
out_proj_kernel(const u16* __restrict__ ao, const float* __restrict__ Wo,
                float* __restrict__ out) {
  const int lane = threadIdx.x;
  const int r = lane & 15, half = lane >> 4;
  const int m0 = (blockIdx.x * 4 + threadIdx.y) * 16;
  const int n0 = blockIdx.y * 16;

  const u16* arow = ao + (size_t)(m0 + r) * DMODEL;
  const float* wrow = Wo + (size_t)(n0 + r) * DMODEL;

  v8f acc = {};
  for (int k0 = 0; k0 < DMODEL; k0 += 32) {
    v16bf a = ld8x2(arow + k0 + half * 16, arow + k0 + half * 16 + 8);
    V16U bw;
    const float* wb = wrow + k0 + half * 16;
#pragma unroll
    for (int j = 0; j < 16; ++j) bw.us[j] = f2bf(wb[j]);
    acc = wmma_bf16(a, bw.bf, acc);
  }
#pragma unroll
  for (int i = 0; i < 8; ++i)
    out[(size_t)(m0 + half * 8 + i) * DMODEL + n0 + r] = acc[i];
}

extern "C" void kernel_launch(void* const* d_in, const int* in_sizes, int n_in,
                              void* d_out, int out_size, void* d_ws, size_t ws_size,
                              hipStream_t stream) {
  const float* x        = (const float*)d_in[0];
  const float* pos_bias = (const float*)d_in[1];
  const float* Wq       = (const float*)d_in[2];
  const float* Wk       = (const float*)d_in[3];
  const float* Wv       = (const float*)d_in[4];
  const float* Wo       = (const float*)d_in[5];
  const int*   mask     = (const int*)d_in[6];
  float* out = (float*)d_out;

  // workspace: q, k, v^T, attn-out -- all bf16, 6.29 MB each (~25 MB total)
  const size_t QKV_ELEMS = (size_t)NB * NHEADS * LL * HDIM;  // 3,145,728
  u16* qd  = (u16*)d_ws;
  u16* kd  = qd + QKV_ELEMS;
  u16* vtd = kd + QKV_ELEMS;
  u16* aod = vtd + QKV_ELEMS;
  (void)ws_size; (void)in_sizes; (void)n_in; (void)out_size;

  // 1) Q/K/V projections: 256 M-tiles x 48 N-tiles each
  qkv_proj_kernel<false><<<dim3(64, 48), dim3(32, 4), 0, stream>>>(x, Wq, qd);
  qkv_proj_kernel<false><<<dim3(64, 48), dim3(32, 4), 0, stream>>>(x, Wk, kd);
  qkv_proj_kernel<true ><<<dim3(64, 48), dim3(32, 4), 0, stream>>>(x, Wv, vtd);

  // 2) attention: 128 query tiles / 4 waves per block, per (h, b)
  attn_kernel<<<dim3(LL / 64, NHEADS, NB), dim3(32, 4), 0, stream>>>(
      qd, kd, vtd, pos_bias, mask, aod);

  // 3) output projection
  out_proj_kernel<<<dim3(64, 48), dim3(32, 4), 0, stream>>>(aod, Wo, out);
}
